// ODEnet_65618510349018
// MI455X (gfx1250) — matrix operand order repaired and checked by reading, not compiled, not run on hardware
//
#include <hip/hip_runtime.h>

typedef __attribute__((ext_vector_type(16))) _Float16 v16h;
typedef __attribute__((ext_vector_type(8)))  float    v8f;

#define B_    16384
#define IN_   512
#define H_    1024
#define OUT_  512
#define NB_   2
#define EPS_  1e-3f
#define NSTEPS 2

#define TILE_M 128
#define TILE_N 128
#define TILE_K 32
#define LDSW   40   // f16 units per LDS row (padded: 80B, conflict-free b128)

union FragU { v16h v; float4 f4[2]; };

// ---------------------------------------------------------------------------
// Fused GEMM:  C[M,N] = epi( preop(A)[M,K] @ Bt[N,K]^T )
//   preop (if p!=null): a = relu(a*p[k] + q[k])          (BN1+relu on input)
//   epi:   c = acc*s[n] + cofs[n]; if relu: c = max(c,0) (bias / BN2+relu)
// A fp32 row-major, Bt f16 row-major [N,K] (pre-transposed weights), C fp32.
// M%128==0, N%128==0, K%32==0.
// ---------------------------------------------------------------------------
__global__ __launch_bounds__(256) void node_gemm_wmma(
    const float* __restrict__ A,
    const _Float16* __restrict__ Bt,
    float* __restrict__ C,
    const float* __restrict__ pre_p,
    const float* __restrict__ pre_q,
    const float* __restrict__ epi_s,
    const float* __restrict__ epi_c,
    int M, int N, int K, int do_relu)
{
    __shared__ alignas(16) _Float16 sA[TILE_M * LDSW];
    __shared__ alignas(16) _Float16 sB[TILE_N * LDSW];

    const int tid  = threadIdx.x;
    const int lane = tid & 31;
    const int wave = tid >> 5;
    const int wm   = wave & 1;    // 0..1 -> 64 rows each
    const int wn   = wave >> 1;   // 0..3 -> 32 cols each
    const int hl   = lane >> 4;   // lane half
    const int lr   = lane & 15;

    const int m0 = blockIdx.y * TILE_M;
    const int n0 = blockIdx.x * TILE_N;

    v8f acc[4][2];
    #pragma unroll
    for (int i = 0; i < 4; ++i)
        #pragma unroll
        for (int j = 0; j < 2; ++j)
            acc[i][j] = (v8f){};

    const int kTiles = K / TILE_K;
    for (int kt = 0; kt < kTiles; ++kt) {
        const int kbase = kt * TILE_K;
        __syncthreads();
        // --- A tile: 128x32 fp32 -> (preop) -> f16 LDS ---
        {
            const int r  = tid >> 3;          // 0..31
            const int c4 = (tid & 7) * 4;     // 0..28
            #pragma unroll
            for (int it = 0; it < 4; ++it) {
                const int row = r + it * 32;
                float4 x = *(const float4*)(A + (size_t)(m0 + row) * K + kbase + c4);
                if (pre_p) {
                    const float4 p = *(const float4*)(pre_p + kbase + c4);
                    const float4 q = *(const float4*)(pre_q + kbase + c4);
                    x.x = fmaxf(x.x * p.x + q.x, 0.0f);
                    x.y = fmaxf(x.y * p.y + q.y, 0.0f);
                    x.z = fmaxf(x.z * p.z + q.z, 0.0f);
                    x.w = fmaxf(x.w * p.w + q.w, 0.0f);
                }
                union { _Float16 h[4]; float2 f2; } cv;
                cv.h[0] = (_Float16)x.x; cv.h[1] = (_Float16)x.y;
                cv.h[2] = (_Float16)x.z; cv.h[3] = (_Float16)x.w;
                *(float2*)(&sA[row * LDSW + c4]) = cv.f2;
            }
        }
        // --- B tile: 128 (N-rows) x 32 (K) f16 ---
        {
            const int r  = tid >> 2;          // 0..63
            const int c8 = (tid & 3) * 8;     // 0,8,16,24
            #pragma unroll
            for (int it = 0; it < 2; ++it) {
                const int row = r + it * 64;
                const uint4 b = *(const uint4*)(Bt + (size_t)(n0 + row) * K + kbase + c8);
                *(uint4*)(&sB[row * LDSW + c8]) = b;
            }
        }
        __syncthreads();

        // --- fragments (layouts per CDNA5 ISA 7.12.2, wave32) ---
        FragU af[4], bf[2];
        #pragma unroll
        for (int i = 0; i < 4; ++i) {
            const int row = wm * 64 + i * 16 + lr;
            const _Float16* base = &sA[row * LDSW + hl * 8];
            af[i].f4[0] = *(const float4*)(base);        // K: hl*8 .. +7
            af[i].f4[1] = *(const float4*)(base + 16);   // K: hl*8+16 .. +7
        }
        #pragma unroll
        for (int j = 0; j < 2; ++j) {
            const int row = wn * 32 + j * 16 + lr;
            const _Float16* base = &sB[row * LDSW + hl * 8];
            bf[j].f4[0] = *(const float4*)(base);        // K: hl*8 .. +7
            bf[j].f4[1] = *(const float4*)(base + 16);   // K: hl*8+16 .. +7
        }
        #pragma unroll
        for (int i = 0; i < 4; ++i)
            #pragma unroll
            for (int j = 0; j < 2; ++j)
                acc[i][j] = __builtin_amdgcn_wmma_f32_16x16x32_f16(
                    false, af[i].v, false, bf[j].v, (short)0, acc[i][j], false, false);
    }

    // --- epilogue ---
    #pragma unroll
    for (int j = 0; j < 2; ++j) {
        const int gn = n0 + wn * 32 + j * 16 + lr;
        const float s = epi_s ? epi_s[gn] : 1.0f;
        const float c = epi_c ? epi_c[gn] : 0.0f;
        #pragma unroll
        for (int i = 0; i < 4; ++i) {
            const int mb = m0 + wm * 64 + i * 16 + hl * 8;
            #pragma unroll
            for (int r = 0; r < 8; ++r) {
                float vv = acc[i][j][r] * s + c;
                if (do_relu) vv = fmaxf(vv, 0.0f);
                C[(size_t)(mb + r) * N + gn] = vv;
            }
        }
    }
}

// --- fp32 [K,N] -> f16 [N,K] (transpose+convert) ---
__global__ void node_convT(const float* __restrict__ W, _Float16* __restrict__ Wt,
                           int K, int N) {
    int idx = blockIdx.x * 256 + threadIdx.x;
    if (idx >= K * N) return;
    int n = idx / K, k = idx - n * K;
    Wt[idx] = (_Float16)W[(size_t)k * N + n];
}

// --- fold BN params: p0/q0 (pre-GEMM1), s1/c1 (post-GEMM1 incl. b1) ---
__global__ void node_prep(const float* __restrict__ g, const float* __restrict__ be,
                          const float* __restrict__ mu, const float* __restrict__ va,
                          const float* __restrict__ b1,
                          float* p0, float* q0, float* s1, float* c1) {
    int i = blockIdx.x * 256 + threadIdx.x;
    if (i >= NB_ * H_) return;
    int b = i / H_, h = i - b * H_;
    int i0 = (b * 2 + 0) * H_ + h;
    int i1 = (b * 2 + 1) * H_ + h;
    float inv0 = g[i0] / sqrtf(va[i0] + EPS_);
    p0[i] = inv0;
    q0[i] = be[i0] - mu[i0] * inv0;
    float inv1 = g[i1] / sqrtf(va[i1] + EPS_);
    s1[i] = inv1;
    c1[i] = (b1[i] - mu[i1]) * inv1 + be[i1];
}

// --- RK4 combine kernels (float4 vectorized) ---
__global__ void node_stage1(const float4* __restrict__ y, const float4* __restrict__ ac,
                            float4* __restrict__ kb, float ck, int n4) {
    int i = blockIdx.x * 256 + threadIdx.x;
    if (i >= n4) return;
    float4 yv = y[i], av = ac[i];
    kb[i] = make_float4(yv.x + ck * av.x, yv.y + ck * av.y,
                        yv.z + ck * av.z, yv.w + ck * av.w);
}
__global__ void node_stage(const float4* __restrict__ y, float4* __restrict__ kb,
                           float4* __restrict__ ac, float wk, float ck, int n4) {
    int i = blockIdx.x * 256 + threadIdx.x;
    if (i >= n4) return;
    float4 kv = kb[i], av = ac[i], yv = y[i];
    ac[i] = make_float4(av.x + wk * kv.x, av.y + wk * kv.y,
                        av.z + wk * kv.z, av.w + wk * kv.w);
    kb[i] = make_float4(yv.x + ck * kv.x, yv.y + ck * kv.y,
                        yv.z + ck * kv.z, yv.w + ck * kv.w);
}
__global__ void node_final(float4* __restrict__ y, const float4* __restrict__ kb,
                           const float4* __restrict__ ac, float h6, int relu, int n4) {
    int i = blockIdx.x * 256 + threadIdx.x;
    if (i >= n4) return;
    float4 yv = y[i], kv = kb[i], av = ac[i];
    float r0 = yv.x + h6 * (av.x + kv.x);
    float r1 = yv.y + h6 * (av.y + kv.y);
    float r2 = yv.z + h6 * (av.z + kv.z);
    float r3 = yv.w + h6 * (av.w + kv.w);
    if (relu) { r0 = fmaxf(r0, 0.f); r1 = fmaxf(r1, 0.f);
                r2 = fmaxf(r2, 0.f); r3 = fmaxf(r3, 0.f); }
    y[i] = make_float4(r0, r1, r2, r3);
}

extern "C" void kernel_launch(void* const* d_in, const int* in_sizes, int n_in,
                              void* d_out, int out_size, void* d_ws, size_t ws_size,
                              hipStream_t stream) {
    const float* inputs = (const float*)d_in[0];
    const float* W_in   = (const float*)d_in[1];
    const float* b_in   = (const float*)d_in[2];
    const float* bn_g   = (const float*)d_in[3];
    const float* bn_b   = (const float*)d_in[4];
    const float* bn_m   = (const float*)d_in[5];
    const float* bn_v   = (const float*)d_in[6];
    const float* W1     = (const float*)d_in[7];
    const float* b1     = (const float*)d_in[8];
    const float* W2     = (const float*)d_in[9];
    const float* b2     = (const float*)d_in[10];
    const float* W_out  = (const float*)d_in[11];
    const float* b_out  = (const float*)d_in[12];

    const size_t BH = (size_t)B_ * H_;
    char* ws = (char*)d_ws;
    float* y    = (float*)ws; ws += BH * 4;
    float* v    = (float*)ws; ws += BH * 4;
    float* kbuf = (float*)ws; ws += BH * 4;
    float* accb = (float*)ws; ws += BH * 4;
    _Float16* WinT  = (_Float16*)ws; ws += (size_t)IN_ * H_ * 2;
    _Float16* W1T   = (_Float16*)ws; ws += (size_t)NB_ * H_ * H_ * 2;
    _Float16* W2T   = (_Float16*)ws; ws += (size_t)NB_ * H_ * H_ * 2;
    _Float16* WoutT = (_Float16*)ws; ws += (size_t)H_ * OUT_ * 2;
    float* p0  = (float*)ws; ws += NB_ * H_ * 4;
    float* q0  = (float*)ws; ws += NB_ * H_ * 4;
    float* s1v = (float*)ws; ws += NB_ * H_ * 4;
    float* c1v = (float*)ws; ws += NB_ * H_ * 4;

    // --- weight convert/transpose + BN folding (cheap, every call) ---
    node_convT<<<(IN_ * H_ + 255) / 256, 256, 0, stream>>>(W_in, WinT, IN_, H_);
    for (int b = 0; b < NB_; ++b) {
        node_convT<<<(H_ * H_ + 255) / 256, 256, 0, stream>>>(
            W1 + (size_t)b * H_ * H_, W1T + (size_t)b * H_ * H_, H_, H_);
        node_convT<<<(H_ * H_ + 255) / 256, 256, 0, stream>>>(
            W2 + (size_t)b * H_ * H_, W2T + (size_t)b * H_ * H_, H_, H_);
    }
    node_convT<<<(H_ * OUT_ + 255) / 256, 256, 0, stream>>>(W_out, WoutT, H_, OUT_);
    node_prep<<<(NB_ * H_ + 255) / 256, 256, 0, stream>>>(
        bn_g, bn_b, bn_m, bn_v, b1, p0, q0, s1v, c1v);

    auto gemm = [&](const float* A, const _Float16* Bt, float* C,
                    const float* pp, const float* pq, const float* es, const float* ec,
                    int M, int N, int K, int relu) {
        dim3 grid(N / TILE_N, M / TILE_M);
        node_gemm_wmma<<<grid, 256, 0, stream>>>(A, Bt, C, pp, pq, es, ec, M, N, K, relu);
    };

    const int n4 = (int)(BH / 4);
    const int eg = (n4 + 255) / 256;
    const float hstep = 1.0f / NSTEPS;

    // input layer: y = X @ W_in + b_in
    gemm(inputs, WinT, y, nullptr, nullptr, nullptr, b_in, B_, H_, IN_, 0);

    for (int b = 0; b < NB_; ++b) {
        const float*    p0b = p0  + b * H_;
        const float*    q0b = q0  + b * H_;
        const float*    s1b = s1v + b * H_;
        const float*    c1b = c1v + b * H_;
        const float*    b2b = b2  + b * H_;
        const _Float16* w1b = W1T + (size_t)b * H_ * H_;
        const _Float16* w2b = W2T + (size_t)b * H_ * H_;

        for (int s = 0; s < NSTEPS; ++s) {
            // k1 -> accb
            gemm(y, w1b, v, p0b, q0b, s1b, c1b, B_, H_, H_, 1);
            gemm(v, w2b, accb, nullptr, nullptr, nullptr, b2b, B_, H_, H_, 0);
            node_stage1<<<eg, 256, 0, stream>>>((float4*)y, (float4*)accb,
                                                (float4*)kbuf, 0.5f * hstep, n4);
            // k2
            gemm(kbuf, w1b, v, p0b, q0b, s1b, c1b, B_, H_, H_, 1);
            gemm(v, w2b, kbuf, nullptr, nullptr, nullptr, b2b, B_, H_, H_, 0);
            node_stage<<<eg, 256, 0, stream>>>((float4*)y, (float4*)kbuf,
                                               (float4*)accb, 2.0f, 0.5f * hstep, n4);
            // k3
            gemm(kbuf, w1b, v, p0b, q0b, s1b, c1b, B_, H_, H_, 1);
            gemm(v, w2b, kbuf, nullptr, nullptr, nullptr, b2b, B_, H_, H_, 0);
            node_stage<<<eg, 256, 0, stream>>>((float4*)y, (float4*)kbuf,
                                               (float4*)accb, 2.0f, hstep, n4);
            // k4 + combine (+relu at end of block)
            gemm(kbuf, w1b, v, p0b, q0b, s1b, c1b, B_, H_, H_, 1);
            gemm(v, w2b, kbuf, nullptr, nullptr, nullptr, b2b, B_, H_, H_, 0);
            node_final<<<eg, 256, 0, stream>>>((float4*)y, (float4*)kbuf,
                                               (float4*)accb, hstep / 6.0f,
                                               (s == NSTEPS - 1) ? 1 : 0, n4);
        }
    }

    // output layer: out = y @ W_out + b_out
    gemm(y, WoutT, (float*)d_out, nullptr, nullptr, nullptr, b_out, B_, OUT_, H_, 0);
}
